// CusparseDynamicLinear_45853070852650
// MI455X (gfx1250) — compile-verified, matrix operand order checked by reading, not run on hardware
//
#include <hip/hip_runtime.h>

// ---------------------------------------------------------------------------
// out[M,N] = data[M,K] @ (mask*weight)[N,K]^T + bias[N]
// fp32 emulated with 3xBF16 WMMA split (hi*hi + hi*lo + lo*hi).
//
// Fast path (needs ws >= (M+N)*K*4 bytes):
//   1) pre-pass splits data and mask*weight into bf16 hi/lo planes in ws
//   2) GEMM stages tiles with GLOBAL_LOAD_ASYNC_TO_LDS_B128 (ASYNCcnt),
//      double-buffered LDS, zero per-iteration conversion VALU.
// Fallback: fused conversion kernel (round-2 design).
// ---------------------------------------------------------------------------

typedef __attribute__((ext_vector_type(16))) __bf16 v16bf;
typedef __attribute__((ext_vector_type(8)))  float  v8f;

constexpr int BM = 128;
constexpr int BN = 128;
constexpr int BK = 32;
constexpr int LDT = 40;     // padded ushort row stride (32 data + 8 pad) = 80B
constexpr int PART = BM * LDT;          // 5120 ushorts per plane tile
constexpr unsigned BUFBYTES = 4 * PART * 2;  // 40960 B per LDS buffer

union FragB16 { v16bf v; uint4 q[2]; };

// pack two bf16 (truncated from f32 bit patterns) with one v_perm_b32
static __device__ inline unsigned pack_hi_pair(unsigned u0, unsigned u1) {
  return __builtin_amdgcn_perm(u1, u0, 0x07060302u);
}

// Split 16 fp32 -> bf16 hi (truncate) + bf16 lo (truncate of residual).
static __device__ inline void cvt_split16(const float* __restrict__ f,
                                          unsigned* __restrict__ hp,
                                          unsigned* __restrict__ lp) {
#pragma unroll
  for (int i = 0; i < 8; ++i) {
    float f0 = f[2 * i], f1 = f[2 * i + 1];
    unsigned u0 = __float_as_uint(f0);
    unsigned u1 = __float_as_uint(f1);
    float r0 = f0 - __uint_as_float(u0 & 0xFFFF0000u);
    float r1 = f1 - __uint_as_float(u1 & 0xFFFF0000u);
    hp[i] = pack_hi_pair(u0, u1);
    lp[i] = pack_hi_pair(__float_as_uint(r0), __float_as_uint(r1));
  }
}

// Load a 16x32 bf16 operand fragment from LDS (ISA 7.12.2 16-bit layout).
static __device__ inline v16bf load_frag_lds(const unsigned short* __restrict__ tile,
                                             int row0, int lrow, int kb) {
  const unsigned short* p = tile + (row0 + lrow) * LDT + kb;
  FragB16 f;
  f.q[0] = *(const uint4*)(p);
  f.q[1] = *(const uint4*)(p + 16);
  return f.v;
}

// ---------------------------------------------------------------------------
// Pre-pass: elementwise split src (optionally * msk) into bf16 hi/lo planes.
// 8 elements per thread.
// ---------------------------------------------------------------------------
__global__ __launch_bounds__(256)
void split_bf16_kernel(const float* __restrict__ src,
                       const float* __restrict__ msk,
                       unsigned short* __restrict__ hi,
                       unsigned short* __restrict__ lo,
                       long n) {
  long i = ((long)blockIdx.x * 256 + threadIdx.x) * 8;
  if (i >= n) return;
  union { float4 q[2]; float f[8]; } v;
  const float4* s4 = (const float4*)(src + i);
  v.q[0] = s4[0]; v.q[1] = s4[1];
  if (msk) {
    union { float4 q[2]; float f[8]; } m;
    const float4* m4 = (const float4*)(msk + i);
    m.q[0] = m4[0]; m.q[1] = m4[1];
#pragma unroll
    for (int j = 0; j < 8; ++j) v.f[j] *= m.f[j];
  }
  unsigned hp[4], lp[4];
#pragma unroll
  for (int j = 0; j < 4; ++j) {
    float f0 = v.f[2 * j], f1 = v.f[2 * j + 1];
    unsigned u0 = __float_as_uint(f0);
    unsigned u1 = __float_as_uint(f1);
    float r0 = f0 - __uint_as_float(u0 & 0xFFFF0000u);
    float r1 = f1 - __uint_as_float(u1 & 0xFFFF0000u);
    hp[j] = pack_hi_pair(u0, u1);
    lp[j] = pack_hi_pair(__float_as_uint(r0), __float_as_uint(r1));
  }
  *(uint4*)(hi + i) = make_uint4(hp[0], hp[1], hp[2], hp[3]);
  *(uint4*)(lo + i) = make_uint4(lp[0], lp[1], lp[2], lp[3]);
}

// ---------------------------------------------------------------------------
// Fast GEMM: pre-split bf16 planes, async LDS staging.
// ---------------------------------------------------------------------------
#define ASYNC_LDS_B128(ldsaddr, voffv, base)                                   \
  asm volatile("global_load_async_to_lds_b128 %0, %1, %2"                      \
               :: "v"(ldsaddr), "v"(voffv), "s"(base) : "memory")

__global__ __launch_bounds__(256)
void gemm_bf16x3_async(const unsigned short* __restrict__ Ahi,
                       const unsigned short* __restrict__ Alo,
                       const unsigned short* __restrict__ Bhi,
                       const unsigned short* __restrict__ Blo,
                       const float* __restrict__ bias,
                       float* __restrict__ out,
                       int M, int N, int K) {
  // [buffer][plane: Ahi,Alo,Bhi,Blo][BM*LDT]
  __shared__ __align__(16) unsigned short S[2][4 * PART];

  const int tid  = threadIdx.x;
  const int lane = tid & 31;
  const int wave = tid >> 5;

  const int m0 = blockIdx.y * BM;
  const int n0 = blockIdx.x * BN;

  const int wm0 = (wave >> 1) * 32;
  const int wn0 = (wave & 1) * 64;

  const int lrow    = lane & 15;
  const int halfsel = lane >> 4;
  const int kb      = halfsel * 8;

  // ---- async staging geometry: per wave 16 rows x 64B, 2 transfers/lane ----
  // transfer t: g = t*32+lane; row = wave*16 + g/4; chunk = g%4 (16B each)
  unsigned voff[2];       // per-lane global byte offset (row*K*2 + chunk*16 + k*2)
  unsigned lds0[2][4];    // buffer-0 LDS byte addresses per transfer per plane
#pragma unroll
  for (int t = 0; t < 2; ++t) {
    const int g    = t * 32 + lane;
    const int row  = wave * 16 + (g >> 2);
    const int chk  = g & 3;
    voff[t] = (unsigned)(row * (K * 2) + chk * 16);
#pragma unroll
    for (int p = 0; p < 4; ++p)
      lds0[t][p] = (unsigned)(uintptr_t)&S[0][p * PART + row * LDT + chk * 8];
  }

  const unsigned long long bAhi = (unsigned long long)(uintptr_t)(Ahi + (size_t)m0 * K);
  const unsigned long long bAlo = (unsigned long long)(uintptr_t)(Alo + (size_t)m0 * K);
  const unsigned long long bBhi = (unsigned long long)(uintptr_t)(Bhi + (size_t)n0 * K);
  const unsigned long long bBlo = (unsigned long long)(uintptr_t)(Blo + (size_t)n0 * K);

  v8f acc[2][4];
#pragma unroll
  for (int mi = 0; mi < 2; ++mi)
#pragma unroll
    for (int ni = 0; ni < 4; ++ni)
      acc[mi][ni] = (v8f){0.f, 0.f, 0.f, 0.f, 0.f, 0.f, 0.f, 0.f};

  const int ksteps = K / BK;

  // ---- prologue: stage k-tile 0 into buffer 0 ----
#pragma unroll
  for (int t = 0; t < 2; ++t) {
    ASYNC_LDS_B128(lds0[t][0], voff[t], bAhi);
    ASYNC_LDS_B128(lds0[t][1], voff[t], bAlo);
    ASYNC_LDS_B128(lds0[t][2], voff[t], bBhi);
    ASYNC_LDS_B128(lds0[t][3], voff[t], bBlo);
  }
  asm volatile("s_wait_asynccnt 0x0" ::: "memory");
  __syncthreads();

  for (int kt = 0; kt < ksteps; ++kt) {
    const int cur   = kt & 1;
    const bool more = (kt + 1) < ksteps;

    // ---- (a) issue next tile's async copies into the other buffer ----
    if (more) {
      const unsigned bufoff = (unsigned)((cur ^ 1) * BUFBYTES);
      voff[0] += BK * 2;
      voff[1] += BK * 2;
#pragma unroll
      for (int t = 0; t < 2; ++t) {
        ASYNC_LDS_B128(lds0[t][0] + bufoff, voff[t], bAhi);
        ASYNC_LDS_B128(lds0[t][1] + bufoff, voff[t], bAlo);
        ASYNC_LDS_B128(lds0[t][2] + bufoff, voff[t], bBhi);
        ASYNC_LDS_B128(lds0[t][3] + bufoff, voff[t], bBlo);
      }
    }

    // ---- (b) fragments + 3xBF16 WMMA from current buffer ----
    const unsigned short* pAhi = &S[cur][0 * PART];
    const unsigned short* pAlo = &S[cur][1 * PART];
    const unsigned short* pBhi = &S[cur][2 * PART];
    const unsigned short* pBlo = &S[cur][3 * PART];

    v16bf ah[2], al[2], bh[4], bl[4];
#pragma unroll
    for (int mi = 0; mi < 2; ++mi) {
      ah[mi] = load_frag_lds(pAhi, wm0 + mi * 16, lrow, kb);
      al[mi] = load_frag_lds(pAlo, wm0 + mi * 16, lrow, kb);
    }
#pragma unroll
    for (int ni = 0; ni < 4; ++ni) {
      bh[ni] = load_frag_lds(pBhi, wn0 + ni * 16, lrow, kb);
      bl[ni] = load_frag_lds(pBlo, wn0 + ni * 16, lrow, kb);
    }
#pragma unroll
    for (int mi = 0; mi < 2; ++mi)
#pragma unroll
      for (int ni = 0; ni < 4; ++ni) {
        acc[mi][ni] = __builtin_amdgcn_wmma_f32_16x16x32_bf16(
            false, ah[mi], false, bh[ni], (short)0, acc[mi][ni], false, false);
        acc[mi][ni] = __builtin_amdgcn_wmma_f32_16x16x32_bf16(
            false, ah[mi], false, bl[ni], (short)0, acc[mi][ni], false, false);
        acc[mi][ni] = __builtin_amdgcn_wmma_f32_16x16x32_bf16(
            false, al[mi], false, bh[ni], (short)0, acc[mi][ni], false, false);
      }

    // ---- (c) wait own async writes (to next buffer), then rendezvous ----
    if (more) asm volatile("s_wait_asynccnt 0x0" ::: "memory");
    __syncthreads();
  }

  // ---- epilogue: C layout (VGPR j -> M = j + 8*halfsel, N = lane&15) ----
#pragma unroll
  for (int ni = 0; ni < 4; ++ni) {
    const int col = n0 + wn0 + ni * 16 + lrow;
    const float bv = bias[col];
#pragma unroll
    for (int mi = 0; mi < 2; ++mi) {
      const int rbase = m0 + wm0 + mi * 16 + halfsel * 8;
#pragma unroll
      for (int j = 0; j < 8; ++j) {
        out[(size_t)(rbase + j) * N + col] = acc[mi][ni][j] + bv;
      }
    }
  }
}

// ---------------------------------------------------------------------------
// Fallback: fused conversion kernel (round-2 design), used if ws too small.
// ---------------------------------------------------------------------------
static __device__ inline void cvt_split_store16(const float* __restrict__ f,
                                                unsigned short* __restrict__ ph,
                                                unsigned short* __restrict__ pl) {
  unsigned hp[8], lp[8];
  cvt_split16(f, hp, lp);
  ((uint4*)ph)[0] = make_uint4(hp[0], hp[1], hp[2], hp[3]);
  ((uint4*)ph)[1] = make_uint4(hp[4], hp[5], hp[6], hp[7]);
  ((uint4*)pl)[0] = make_uint4(lp[0], lp[1], lp[2], lp[3]);
  ((uint4*)pl)[1] = make_uint4(lp[4], lp[5], lp[6], lp[7]);
}

__global__ __launch_bounds__(256)
void masked_linear_fused(const float* __restrict__ data,
                         const float* __restrict__ weight,
                         const float* __restrict__ mask,
                         const float* __restrict__ bias,
                         float* __restrict__ out,
                         int M, int N, int K) {
  __shared__ __align__(16) unsigned short sAhi[2][PART];
  __shared__ __align__(16) unsigned short sAlo[2][PART];
  __shared__ __align__(16) unsigned short sBhi[2][PART];
  __shared__ __align__(16) unsigned short sBlo[2][PART];

  const int tid  = threadIdx.x;
  const int lane = tid & 31;
  const int wave = tid >> 5;
  const int m0 = blockIdx.y * BM;
  const int n0 = blockIdx.x * BN;
  const int wm0 = (wave >> 1) * 32;
  const int wn0 = (wave & 1) * 64;
  const int srow = tid >> 1;
  const int scs  = (tid & 1) * 16;
  const int soff = srow * LDT + scs;
  const size_t aoff = (size_t)(m0 + srow) * K + scs;
  const size_t boff = (size_t)(n0 + srow) * K + scs;
  const int lrow    = lane & 15;
  const int halfsel = lane >> 4;
  const int kbf     = halfsel * 8;

  v8f acc[2][4];
#pragma unroll
  for (int mi = 0; mi < 2; ++mi)
#pragma unroll
    for (int ni = 0; ni < 4; ++ni)
      acc[mi][ni] = (v8f){0.f, 0.f, 0.f, 0.f, 0.f, 0.f, 0.f, 0.f};

  const int ksteps = K / BK;
  float af[16], wf[16];
  {
    const float4* ga = (const float4*)(data + aoff);
    const float4* gw = (const float4*)(weight + boff);
    const float4* gm = (const float4*)(mask + boff);
    union { float4 q[4]; float f[16]; } A, W, Mk;
    A.q[0] = ga[0]; A.q[1] = ga[1]; A.q[2] = ga[2]; A.q[3] = ga[3];
    W.q[0] = gw[0]; W.q[1] = gw[1]; W.q[2] = gw[2]; W.q[3] = gw[3];
    Mk.q[0] = gm[0]; Mk.q[1] = gm[1]; Mk.q[2] = gm[2]; Mk.q[3] = gm[3];
#pragma unroll
    for (int i = 0; i < 16; ++i) { af[i] = A.f[i]; wf[i] = W.f[i] * Mk.f[i]; }
    cvt_split_store16(af, &sAhi[0][soff], &sAlo[0][soff]);
    cvt_split_store16(wf, &sBhi[0][soff], &sBlo[0][soff]);
  }
  __syncthreads();

  for (int kt = 0; kt < ksteps; ++kt) {
    const int cur  = kt & 1;
    const bool more = (kt + 1) < ksteps;
    if (more) {
      const int kc = (kt + 1) * BK;
      const float4* ga = (const float4*)(data + aoff + kc);
      const float4* gw = (const float4*)(weight + boff + kc);
      const float4* gm = (const float4*)(mask + boff + kc);
      union { float4 q[4]; float f[16]; } A, W, Mk;
      A.q[0] = ga[0]; A.q[1] = ga[1]; A.q[2] = ga[2]; A.q[3] = ga[3];
      W.q[0] = gw[0]; W.q[1] = gw[1]; W.q[2] = gw[2]; W.q[3] = gw[3];
      Mk.q[0] = gm[0]; Mk.q[1] = gm[1]; Mk.q[2] = gm[2]; Mk.q[3] = gm[3];
#pragma unroll
      for (int i = 0; i < 16; ++i) { af[i] = A.f[i]; wf[i] = W.f[i] * Mk.f[i]; }
    }

    v16bf ah[2], al[2], bh[4], bl[4];
#pragma unroll
    for (int mi = 0; mi < 2; ++mi) {
      ah[mi] = load_frag_lds(sAhi[cur], wm0 + mi * 16, lrow, kbf);
      al[mi] = load_frag_lds(sAlo[cur], wm0 + mi * 16, lrow, kbf);
    }
#pragma unroll
    for (int ni = 0; ni < 4; ++ni) {
      bh[ni] = load_frag_lds(sBhi[cur], wn0 + ni * 16, lrow, kbf);
      bl[ni] = load_frag_lds(sBlo[cur], wn0 + ni * 16, lrow, kbf);
    }
#pragma unroll
    for (int mi = 0; mi < 2; ++mi)
#pragma unroll
      for (int ni = 0; ni < 4; ++ni) {
        acc[mi][ni] = __builtin_amdgcn_wmma_f32_16x16x32_bf16(
            false, ah[mi], false, bh[ni], (short)0, acc[mi][ni], false, false);
        acc[mi][ni] = __builtin_amdgcn_wmma_f32_16x16x32_bf16(
            false, ah[mi], false, bl[ni], (short)0, acc[mi][ni], false, false);
        acc[mi][ni] = __builtin_amdgcn_wmma_f32_16x16x32_bf16(
            false, al[mi], false, bh[ni], (short)0, acc[mi][ni], false, false);
      }

    if (more) {
      const int nxt = cur ^ 1;
      cvt_split_store16(af, &sAhi[nxt][soff], &sAlo[nxt][soff]);
      cvt_split_store16(wf, &sBhi[nxt][soff], &sBlo[nxt][soff]);
    }
    __syncthreads();
  }

#pragma unroll
  for (int ni = 0; ni < 4; ++ni) {
    const int col = n0 + wn0 + ni * 16 + lrow;
    const float bv = bias[col];
#pragma unroll
    for (int mi = 0; mi < 2; ++mi) {
      const int rbase = m0 + wm0 + mi * 16 + halfsel * 8;
#pragma unroll
      for (int j = 0; j < 8; ++j) {
        out[(size_t)(rbase + j) * N + col] = acc[mi][ni][j] + bv;
      }
    }
  }
}

// ---------------------------------------------------------------------------
extern "C" void kernel_launch(void* const* d_in, const int* in_sizes, int n_in,
                              void* d_out, int out_size, void* d_ws, size_t ws_size,
                              hipStream_t stream) {
  const float* data   = (const float*)d_in[0];
  const float* weight = (const float*)d_in[1];
  const float* mask   = (const float*)d_in[2];
  const float* bias   = (const float*)d_in[3];
  float* out = (float*)d_out;

  const int N = in_sizes[3];              // bias length
  const int K = in_sizes[1] / N;          // weight is [N,K]
  const int M = in_sizes[0] / K;          // data is [M,K]

  const size_t MK = (size_t)M * K;
  const size_t NK = (size_t)N * K;
  const size_t need = (MK + NK) * 4;      // bf16 hi+lo planes for A and B

  dim3 grid((unsigned)(N / BN), (unsigned)(M / BM));

  if (ws_size >= need) {
    unsigned short* wsb = (unsigned short*)d_ws;
    unsigned short* Bhi = wsb;
    unsigned short* Blo = wsb + NK;
    unsigned short* Ahi = wsb + 2 * NK;
    unsigned short* Alo = Ahi + MK;

    split_bf16_kernel<<<(unsigned)(MK / 2048), 256, 0, stream>>>(
        data, nullptr, Ahi, Alo, (long)MK);
    split_bf16_kernel<<<(unsigned)(NK / 2048), 256, 0, stream>>>(
        weight, mask, Bhi, Blo, (long)NK);
    gemm_bf16x3_async<<<grid, 256, 0, stream>>>(Ahi, Alo, Bhi, Blo, bias, out,
                                                M, N, K);
  } else {
    masked_linear_fused<<<grid, 256, 0, stream>>>(data, weight, mask, bias,
                                                  out, M, N, K);
  }
}
